// SwitchTransformerLayer_13984413516053
// MI455X (gfx1250) — compile-verified, matrix operand order checked by reading, not run on hardware
//
#include <hip/hip_runtime.h>
#include <hip/hip_bf16.h>
#include <stdint.h>

// SwitchTransformer layer for gfx1250 (MI455X), bf16 WMMA with fp32 accumulate.
// B=4 S=1024 D=1024 H=16 DK=64 E=8 DF=4096, N=4096 tokens.

#define D_MODEL 1024
#define N_TOK   4096
#define N_HEAD  16
#define D_HEAD  64
#define SEQ     1024
#define BATCH   4
#define N_EXP   8
#define D_FF    4096

#define USE_ASYNC_LDS 1

typedef __attribute__((ext_vector_type(16))) __bf16 v16bf;
typedef __attribute__((ext_vector_type(8)))  __bf16 v8bf;
typedef __attribute__((ext_vector_type(8)))  float  v8f;
typedef __attribute__((ext_vector_type(4)))  float  v4f;

// Build a 16x32 bf16 A/B fragment from LDS: two 16-byte chunks per lane
// (K-halves split across lane halves per the CDNA5 WMMA VGPR layout).
static __device__ __forceinline__ v16bf frag_from_lds(const __bf16* lo, const __bf16* hi) {
  v8bf a = *(const v8bf*)lo;
  v8bf b = *(const v8bf*)hi;
  v16bf r;
#pragma unroll
  for (int i = 0; i < 8; ++i) { r[i] = a[i]; r[8 + i] = b[i]; }
  return r;
}

static __device__ __forceinline__ v8f wmma_bf16(v16bf a, v16bf b, v8f c) {
  // (neg_a, A, neg_b, B, c_mod, C, reuse_a, reuse_b)
  return __builtin_amdgcn_wmma_f32_16x16x32_bf16(false, a, false, b, (short)0, c, false, false);
}

// CDNA5 async global->LDS copy (ASYNCcnt-tracked), 8 bytes per lane.
static __device__ __forceinline__ void async_copy_b64(uint32_t lds_off, const void* gaddr) {
#if USE_ASYNC_LDS
  asm volatile("global_load_async_to_lds_b64 %0, %1, off"
               :: "v"(lds_off), "v"((uint64_t)(uintptr_t)gaddr)
               : "memory");
#else
  (void)lds_off; (void)gaddr;
#endif
}

static __device__ __forceinline__ void wait_asynccnt0() {
#if USE_ASYNC_LDS
  asm volatile("s_wait_asynccnt 0x0" ::: "memory");
#endif
}

// ---------------------------------------------------------------- layernorm
__global__ __launch_bounds__(256) void ln_kernel(const float* __restrict__ x,
                                                 const float* __restrict__ g,
                                                 const float* __restrict__ b,
                                                 float* __restrict__ y) {
  const int row = blockIdx.x, t = threadIdx.x;
  const float* xr = x + (size_t)row * D_MODEL;
  __shared__ float red[256];
  float s = 0.f;
  for (int i = t; i < D_MODEL; i += 256) s += xr[i];
  red[t] = s; __syncthreads();
  for (int o = 128; o > 0; o >>= 1) { if (t < o) red[t] += red[t + o]; __syncthreads(); }
  const float mu = red[0] * (1.f / D_MODEL);
  __syncthreads();
  float v = 0.f;
  for (int i = t; i < D_MODEL; i += 256) { float d = xr[i] - mu; v += d * d; }
  red[t] = v; __syncthreads();
  for (int o = 128; o > 0; o >>= 1) { if (t < o) red[t] += red[t + o]; __syncthreads(); }
  const float rs = rsqrtf(red[0] * (1.f / D_MODEL) + 1e-5f);
  float* yr = y + (size_t)row * D_MODEL;
  for (int i = t; i < D_MODEL; i += 256) yr[i] = (xr[i] - mu) * rs * g[i] + b[i];
}

// ------------------------------------------------- dense GEMM: C = A@B + bias
// Tile 128x64, BK=32, 8 waves, double-buffered LDS + software pipeline:
// global loads for tile k+1 are in flight while WMMAs run on tile k.
__global__ __launch_bounds__(256) void gemm_bias_kernel(const float* __restrict__ A,
                                                        const float* __restrict__ B,
                                                        const float* __restrict__ bias,
                                                        float* __restrict__ C,
                                                        int M, int N, int K) {
  __shared__ __bf16 sA[2][128 * 32];   // [m][k]
  __shared__ __bf16 sB[2][64 * 32];    // [n][k]
  const int t = threadIdx.x;
  const int wv = t >> 5, lane = t & 31, hf = lane >> 4, ln16 = lane & 15;
  const int rowBase = blockIdx.y * 128, colBase = blockIdx.x * 64;

  v4f ra[4], rb[2];
  auto load_regs = [&](int k0) {
#pragma unroll
    for (int it = 0; it < 4; ++it) {
      const int i = t + it * 256, r = i >> 3, kk = (i & 7) << 2;
      ra[it] = *(const v4f*)(A + (size_t)(rowBase + r) * K + k0 + kk);
    }
#pragma unroll
    for (int it = 0; it < 2; ++it) {
      const int i = t + it * 256, kr = i >> 4, nn = (i & 15) << 2;
      rb[it] = *(const v4f*)(B + (size_t)(k0 + kr) * N + colBase + nn);
      if (k0 + 32 < K)  // prefetch the weight tile after next
        __builtin_prefetch(B + (size_t)(k0 + 32 + kr) * N + colBase + nn, 0, 1);
    }
  };
  auto store_lds = [&](int buf) {
#pragma unroll
    for (int it = 0; it < 4; ++it) {
      const int i = t + it * 256, r = i >> 3, kk = (i & 7) << 2;
      __bf16* d = &sA[buf][r * 32 + kk];
#pragma unroll
      for (int j = 0; j < 4; ++j) d[j] = (__bf16)ra[it][j];
    }
#pragma unroll
    for (int it = 0; it < 2; ++it) {
      const int i = t + it * 256, kr = i >> 4, nn = (i & 15) << 2;
#pragma unroll
      for (int j = 0; j < 4; ++j) sB[buf][(nn + j) * 32 + kr] = (__bf16)rb[it][j];
    }
  };

  v8f acc[4] = {};
  load_regs(0);
  store_lds(0);
  const int nk = K / 32;
  for (int kt = 0; kt < nk; ++kt) {
    __syncthreads();
    if (kt + 1 < nk) load_regs((kt + 1) * 32);   // overlap with WMMAs below
    const int buf = kt & 1;
    const int arow = wv * 16 + ln16;
    v16bf af = frag_from_lds(&sA[buf][arow * 32 + hf * 8],
                             &sA[buf][arow * 32 + 16 + hf * 8]);
    // Load all B fragments first, then issue the WMMAs back-to-back.
    v16bf bfr[4];
#pragma unroll
    for (int f = 0; f < 4; ++f) {
      const int bcol = f * 16 + ln16;
      bfr[f] = frag_from_lds(&sB[buf][bcol * 32 + hf * 8],
                             &sB[buf][bcol * 32 + 16 + hf * 8]);
    }
#pragma unroll
    for (int f = 0; f < 4; ++f) acc[f] = wmma_bf16(af, bfr[f], acc[f]);
    if (kt + 1 < nk) store_lds((kt + 1) & 1);
  }
#pragma unroll
  for (int f = 0; f < 4; ++f) {
    const int col = colBase + f * 16 + ln16;
    const float bz = bias ? bias[col] : 0.f;
#pragma unroll
    for (int r = 0; r < 8; ++r) {
      const int row = rowBase + wv * 16 + r + hf * 8;  // C-frag: M = r + 8*half
      C[(size_t)row * N + col] = acc[f][r] + bz;
    }
  }
}

// ------------------------------------------------------------- flash attention
// One block = 4 waves x 16-query tiles (64 queries), shared K/V staging.
__global__ __launch_bounds__(128) void attn_kernel(const float* __restrict__ Q,
                                                   const float* __restrict__ K,
                                                   const float* __restrict__ V,
                                                   float* __restrict__ O) {
  __shared__ __bf16 sK[32 * 64];      // [key][d]
  __shared__ __bf16 sVt[64 * 32];     // [d][key]
  __shared__ __bf16 sP[4][16 * 32];   // per-wave probs bounce [row][key]
  const int t = threadIdx.x, wv = t >> 5, lane = t & 31, hf = lane >> 4, ln16 = lane & 15;
  const int bh = blockIdx.y, bidx = bh >> 4, hidx = bh & 15;
  const int qrow_l = blockIdx.x * 64 + wv * 16 + ln16;  // A-frag row (both halves)
  const float* qp = Q + ((size_t)bidx * SEQ + qrow_l) * D_MODEL + hidx * D_HEAD;
  v16bf qf[2];
#pragma unroll
  for (int s = 0; s < 2; ++s) {
    const float* p0 = qp + s * 32 + hf * 8;
    const float* p1 = p0 + 16;
    v4f a0 = *(const v4f*)p0, a1 = *(const v4f*)(p0 + 4);
    v4f b0 = *(const v4f*)p1, b1 = *(const v4f*)(p1 + 4);
    v16bf f;
#pragma unroll
    for (int i = 0; i < 4; ++i) {
      f[i]      = (__bf16)(a0[i] * 0.125f);
      f[4 + i]  = (__bf16)(a1[i] * 0.125f);
      f[8 + i]  = (__bf16)(b0[i] * 0.125f);
      f[12 + i] = (__bf16)(b1[i] * 0.125f);
    }
    qf[s] = f;
  }
  v8f oacc[4] = {};
  float mrow[8], lrow[8];
#pragma unroll
  for (int r = 0; r < 8; ++r) { mrow[r] = -1e30f; lrow[r] = 0.f; }
  const float* kbase = K + (size_t)bidx * SEQ * D_MODEL + hidx * D_HEAD;
  const float* vbase = V + (size_t)bidx * SEQ * D_MODEL + hidx * D_HEAD;

  for (int kt = 0; kt < SEQ / 32; ++kt) {
#pragma unroll
    for (int it = 0; it < 4; ++it) {
      const int i = t + it * 128;
      const int kk = i >> 4, dd = (i & 15) << 2;
      v4f kv = *(const v4f*)(kbase + (size_t)(kt * 32 + kk) * D_MODEL + dd);
      v4f vv = *(const v4f*)(vbase + (size_t)(kt * 32 + kk) * D_MODEL + dd);
      __bf16* kd = &sK[kk * 64 + dd];
#pragma unroll
      for (int j = 0; j < 4; ++j) { kd[j] = (__bf16)kv[j]; sVt[(dd + j) * 32 + kk] = (__bf16)vv[j]; }
    }
    __syncthreads();
    // scores: S[16q x 32k] = Q(16x64) @ K^T(64x32); load all K frags first.
    v16bf kfr[2][2];
#pragma unroll
    for (int s = 0; s < 2; ++s)
#pragma unroll
      for (int fk = 0; fk < 2; ++fk) {
        const __bf16* base = &sK[(fk * 16 + ln16) * 64 + s * 32 + hf * 8];
        kfr[s][fk] = frag_from_lds(base, base + 16);
      }
    v8f sacc[2] = {};
#pragma unroll
    for (int s = 0; s < 2; ++s)
#pragma unroll
      for (int fk = 0; fk < 2; ++fk)
        sacc[fk] = wmma_bf16(qf[s], kfr[s][fk], sacc[fk]);
#pragma unroll
    for (int r = 0; r < 8; ++r) {
      float tmax = fmaxf(sacc[0][r], sacc[1][r]);
#pragma unroll
      for (int m = 1; m < 16; m <<= 1) tmax = fmaxf(tmax, __shfl_xor(tmax, m, 32));
      const float mnew = fmaxf(mrow[r], tmax);
      const float scale = __expf(mrow[r] - mnew);
      const float p0 = __expf(sacc[0][r] - mnew);
      const float p1 = __expf(sacc[1][r] - mnew);
      float ps = p0 + p1;
#pragma unroll
      for (int m = 1; m < 16; m <<= 1) ps += __shfl_xor(ps, m, 32);
      lrow[r] = lrow[r] * scale + ps;
      mrow[r] = mnew;
#pragma unroll
      for (int f = 0; f < 4; ++f) oacc[f][r] *= scale;
      const int prow = r + hf * 8;
      sP[wv][prow * 32 + ln16]      = (__bf16)p0;
      sP[wv][prow * 32 + 16 + ln16] = (__bf16)p1;
    }
    // O += P(16x32) @ V(32x64); load all V frags first.
    v16bf pf = frag_from_lds(&sP[wv][ln16 * 32 + hf * 8], &sP[wv][ln16 * 32 + 16 + hf * 8]);
    v16bf vfr[4];
#pragma unroll
    for (int f = 0; f < 4; ++f) {
      const __bf16* base = &sVt[(f * 16 + ln16) * 32 + hf * 8];
      vfr[f] = frag_from_lds(base, base + 16);
    }
#pragma unroll
    for (int f = 0; f < 4; ++f) oacc[f] = wmma_bf16(pf, vfr[f], oacc[f]);
    __syncthreads();
  }
#pragma unroll
  for (int r = 0; r < 8; ++r) {
    const float inv = 1.f / lrow[r];
    const int row = blockIdx.x * 64 + wv * 16 + r + hf * 8;
    float* op = O + ((size_t)bidx * SEQ + row) * D_MODEL + hidx * D_HEAD;
#pragma unroll
    for (int f = 0; f < 4; ++f) op[f * 16 + ln16] = oacc[f][r] * inv;
  }
}

// --------------------------------------------------------------- residual add
__global__ __launch_bounds__(256) void add_kernel(const float* __restrict__ a,
                                                  const float* __restrict__ b,
                                                  float* __restrict__ c) {
  const size_t i = (size_t)blockIdx.x * 256 + threadIdx.x;
  const v4f* av = (const v4f*)a;
  const v4f* bv = (const v4f*)b;
  ((v4f*)c)[i] = av[i] + bv[i];
}

// --------------------------------------------------------------------- router
__global__ __launch_bounds__(256) void gate_kernel(const float* __restrict__ Xln,
                                                   const float* __restrict__ gW,
                                                   const float* __restrict__ gb,
                                                   float* __restrict__ probs,
                                                   float* __restrict__ mval,
                                                   float* __restrict__ pi_tok,
                                                   int* __restrict__ eidx,
                                                   int* __restrict__ counts) {
  const int wv = threadIdx.x >> 5, lane = threadIdx.x & 31;
  const int tok = blockIdx.x * 8 + wv;
  const float* xr = Xln + (size_t)tok * D_MODEL;
  float acc[N_EXP] = {};
  for (int d = lane; d < D_MODEL; d += 32) {
    const float xv = xr[d];
#pragma unroll
    for (int e = 0; e < N_EXP; ++e) acc[e] += xv * gW[d * N_EXP + e];
  }
#pragma unroll
  for (int e = 0; e < N_EXP; ++e) {
#pragma unroll
    for (int m = 1; m < 32; m <<= 1) acc[e] += __shfl_xor(acc[e], m, 32);
  }
  if (lane == 0) {
    float mx = -1e30f;
#pragma unroll
    for (int e = 0; e < N_EXP; ++e) { acc[e] += gb[e]; mx = fmaxf(mx, acc[e]); }
    float s = 0.f;
#pragma unroll
    for (int e = 0; e < N_EXP; ++e) { acc[e] = __expf(acc[e] - mx); s += acc[e]; }
    const float inv = 1.f / s;
    float best = -1.f, mean = 0.f; int bi = 0;
#pragma unroll
    for (int e = 0; e < N_EXP; ++e) {
      const float p = acc[e] * inv;
      probs[(size_t)tok * N_EXP + e] = p;
      mean += p;
      if (p > best) { best = p; bi = e; }
    }
    mval[tok] = best;
    pi_tok[tok] = mean * (1.f / N_EXP);
    eidx[tok] = bi;
    atomicAdd(&counts[bi], 1);
  }
}

__global__ void zero_small_kernel(int* counts, int* cursor) {
  const int t = threadIdx.x;
  if (t < N_EXP) { counts[t] = 0; cursor[t] = 0; }
}

__global__ void scan_kernel(const int* __restrict__ counts, int* __restrict__ basep) {
  if (threadIdx.x == 0) {
    int s = 0;
    for (int e = 0; e < N_EXP; ++e) { basep[e] = s; s += counts[e]; }
  }
}

__global__ __launch_bounds__(256) void perm_kernel(const int* __restrict__ eidx,
                                                   const int* __restrict__ basep,
                                                   int* __restrict__ cursor,
                                                   int* __restrict__ perm) {
  const int t = blockIdx.x * 256 + threadIdx.x;
  if (t < N_TOK) {
    const int e = eidx[t];
    const int pos = atomicAdd(&cursor[e], 1);
    perm[basep[e] + pos] = t;
  }
}

// --------------------------------------- grouped MoE GEMM 1: H = relu(X@W1+b1)
// 128 gathered tokens x 64 cols, K=1024; double-buffered software pipeline.
__global__ __launch_bounds__(256) void moe_ffn1_kernel(const float* __restrict__ Xln,
                                                       const float* __restrict__ W1,
                                                       const float* __restrict__ b1,
                                                       const int* __restrict__ perm,
                                                       const int* __restrict__ basep,
                                                       const int* __restrict__ counts,
                                                       __bf16* __restrict__ Hbuf) {
  const int e = blockIdx.y >> 5, tile = blockIdx.y & 31;
  const int cnt = counts[e];
  if (tile * 128 >= cnt) return;
  __shared__ __bf16 sA[2][128 * 32];
  __shared__ __bf16 sB[2][64 * 32];
  __shared__ int s_tok[128];
  const int t = threadIdx.x, wv = t >> 5, lane = t & 31, hf = lane >> 4, ln16 = lane & 15;
  if (t < 128) {
    const int i = tile * 128 + t;
    s_tok[t] = (i < cnt) ? perm[basep[e] + i] : -1;
  }
  __syncthreads();
  const int nbase = blockIdx.x * 64;
  const float* Be = W1 + (size_t)e * D_MODEL * D_FF;

  v4f ra[4], rb[2];
  auto load_regs = [&](int k0) {
#pragma unroll
    for (int it = 0; it < 4; ++it) {
      const int i = t + it * 256, r = i >> 3, kk = (i & 7) << 2;
      const int tok = s_tok[r];
      v4f z = {};
      ra[it] = (tok >= 0) ? *(const v4f*)(Xln + (size_t)tok * D_MODEL + k0 + kk) : z;
    }
#pragma unroll
    for (int it = 0; it < 2; ++it) {
      const int i = t + it * 256, kr = i >> 4, nn = (i & 15) << 2;
      rb[it] = *(const v4f*)(Be + (size_t)(k0 + kr) * D_FF + nbase + nn);
      if (k0 + 32 < D_MODEL)
        __builtin_prefetch(Be + (size_t)(k0 + 32 + kr) * D_FF + nbase + nn, 0, 1);
    }
  };
  auto store_lds = [&](int buf) {
#pragma unroll
    for (int it = 0; it < 4; ++it) {
      const int i = t + it * 256, r = i >> 3, kk = (i & 7) << 2;
      __bf16* d = &sA[buf][r * 32 + kk];
#pragma unroll
      for (int j = 0; j < 4; ++j) d[j] = (__bf16)ra[it][j];
    }
#pragma unroll
    for (int it = 0; it < 2; ++it) {
      const int i = t + it * 256, kr = i >> 4, nn = (i & 15) << 2;
#pragma unroll
      for (int j = 0; j < 4; ++j) sB[buf][(nn + j) * 32 + kr] = (__bf16)rb[it][j];
    }
  };

  v8f acc[4] = {};
  load_regs(0);
  store_lds(0);
  const int nk = D_MODEL / 32;
  for (int kt = 0; kt < nk; ++kt) {
    __syncthreads();
    if (kt + 1 < nk) load_regs((kt + 1) * 32);
    const int buf = kt & 1;
    const int arow = wv * 16 + ln16;
    v16bf af = frag_from_lds(&sA[buf][arow * 32 + hf * 8],
                             &sA[buf][arow * 32 + 16 + hf * 8]);
    v16bf bfr[4];
#pragma unroll
    for (int f = 0; f < 4; ++f) {
      const int bcol = f * 16 + ln16;
      bfr[f] = frag_from_lds(&sB[buf][bcol * 32 + hf * 8],
                             &sB[buf][bcol * 32 + 16 + hf * 8]);
    }
#pragma unroll
    for (int f = 0; f < 4; ++f) acc[f] = wmma_bf16(af, bfr[f], acc[f]);
    if (kt + 1 < nk) store_lds((kt + 1) & 1);
  }
#pragma unroll
  for (int f = 0; f < 4; ++f) {
    const int col = nbase + f * 16 + ln16;
    const float bz = b1[(size_t)e * D_FF + col];
#pragma unroll
    for (int r = 0; r < 8; ++r) {
      const int rl = wv * 16 + r + hf * 8;
      const int tok = s_tok[rl];
      if (tok >= 0) Hbuf[(size_t)tok * D_FF + col] = (__bf16)fmaxf(acc[f][r] + bz, 0.f);
    }
  }
}

// --------- grouped MoE GEMM 2 + epilogue: out = xres + mval*(H@W2 + b2), K=4096
// A-operand (bf16 hidden) streams via CDNA5 async global->LDS DMA into the
// next buffer while WMMAs run on the current one. Every thread issues exactly
// 4 async ops per stage (padding rows go to a trash slot so pre-zeroed padding
// rows stay zero), and waits ASYNCcnt==0 before the barrier that publishes them.
__global__ __launch_bounds__(256) void moe_ffn2_kernel(const __bf16* __restrict__ Hbuf,
                                                       const float* __restrict__ W2,
                                                       const float* __restrict__ b2,
                                                       const int* __restrict__ perm,
                                                       const int* __restrict__ basep,
                                                       const int* __restrict__ counts,
                                                       const float* __restrict__ xres,
                                                       const float* __restrict__ mval,
                                                       float* __restrict__ out) {
  const int e = blockIdx.y >> 5, tile = blockIdx.y & 31;
  const int cnt = counts[e];
  if (tile * 128 >= cnt) return;
  __shared__ __bf16 sA[2][128 * 32];
  __shared__ __bf16 sB[2][64 * 32];
  __shared__ __bf16 sTrash[256 * 4];
  __shared__ int s_tok[128];
  const int t = threadIdx.x, wv = t >> 5, lane = t & 31, hf = lane >> 4, ln16 = lane & 15;
  if (t < 128) {
    const int i = tile * 128 + t;
    s_tok[t] = (i < cnt) ? perm[basep[e] + i] : -1;
  }
  // Pre-zero both A buffers so padding rows contribute zero in every K-step.
  for (int i = t; i < (2 * 128 * 32 * 2) / 8; i += 256) ((uint64_t*)sA)[i] = 0ull;
  __syncthreads();
  const int nbase = blockIdx.x * 64;
  const float* Be = W2 + (size_t)e * D_FF * D_MODEL;
  const uint32_t trashOff = (uint32_t)(uintptr_t)&sTrash[t * 4];

  auto stageA_async = [&](int k0, int buf) {
#pragma unroll
    for (int it = 0; it < 4; ++it) {
      const int i = t + it * 256, r = i >> 3, c = i & 7;
      const int tok = s_tok[r];
      const __bf16* g = Hbuf + (size_t)(tok >= 0 ? tok : 0) * D_FF + k0 + c * 4;
      const uint32_t loff = (tok >= 0)
          ? (uint32_t)(uintptr_t)&sA[buf][r * 32 + c * 4] : trashOff;
      async_copy_b64(loff, g);
    }
  };
  v4f rb[2];
  auto loadB_regs = [&](int k0) {
#pragma unroll
    for (int it = 0; it < 2; ++it) {
      const int i = t + it * 256, kr = i >> 4, nn = (i & 15) << 2;
      rb[it] = *(const v4f*)(Be + (size_t)(k0 + kr) * D_MODEL + nbase + nn);
      if (k0 + 32 < D_FF)
        __builtin_prefetch(Be + (size_t)(k0 + 32 + kr) * D_MODEL + nbase + nn, 0, 1);
    }
  };
  auto storeB_lds = [&](int buf) {
#pragma unroll
    for (int it = 0; it < 2; ++it) {
      const int i = t + it * 256, kr = i >> 4, nn = (i & 15) << 2;
#pragma unroll
      for (int j = 0; j < 4; ++j) sB[buf][(nn + j) * 32 + kr] = (__bf16)rb[it][j];
    }
  };

  v8f acc[4] = {};
  stageA_async(0, 0);
  loadB_regs(0);
  storeB_lds(0);
  wait_asynccnt0();
  const int nk = D_FF / 32;
  for (int kt = 0; kt < nk; ++kt) {
    __syncthreads();  // publishes tile kt (async A + B stores) to all waves
    if (kt + 1 < nk) { stageA_async((kt + 1) * 32, (kt + 1) & 1); loadB_regs((kt + 1) * 32); }
    const int buf = kt & 1;
    const int arow = wv * 16 + ln16;
    v16bf af = frag_from_lds(&sA[buf][arow * 32 + hf * 8],
                             &sA[buf][arow * 32 + 16 + hf * 8]);
    v16bf bfr[4];
#pragma unroll
    for (int f = 0; f < 4; ++f) {
      const int bcol = f * 16 + ln16;
      bfr[f] = frag_from_lds(&sB[buf][bcol * 32 + hf * 8],
                             &sB[buf][bcol * 32 + 16 + hf * 8]);
    }
#pragma unroll
    for (int f = 0; f < 4; ++f) acc[f] = wmma_bf16(af, bfr[f], acc[f]);
    if (kt + 1 < nk) storeB_lds((kt + 1) & 1);
    wait_asynccnt0();  // own async for tile kt+1 landed before next barrier
  }
#pragma unroll
  for (int f = 0; f < 4; ++f) {
    const int col = nbase + f * 16 + ln16;
    const float bz = b2[(size_t)e * D_MODEL + col];
#pragma unroll
    for (int r = 0; r < 8; ++r) {
      const int rl = wv * 16 + r + hf * 8;
      const int tok = s_tok[rl];
      if (tok >= 0) {
        out[(size_t)tok * D_MODEL + col] =
            xres[(size_t)tok * D_MODEL + col] + mval[tok] * (acc[f][r] + bz);
      }
    }
  }
}

// ------------------------------------------------------- load-balance scalar
__global__ void lbl_kernel(const int* __restrict__ counts,
                           const float* __restrict__ pi_tok,
                           float* __restrict__ out) {
  if (threadIdx.x == 0 && blockIdx.x == 0) {
    float s = 0.f;
    for (int e = 0; e < N_EXP; ++e)
      s += ((float)counts[e] * (1.f / N_TOK)) * pi_tok[e];
    out[(size_t)N_TOK * D_MODEL] = 0.01f * (float)N_EXP * s;
  }
}

// ---------------------------------------------------------------------- host
extern "C" void kernel_launch(void* const* d_in, const int* in_sizes, int n_in,
                              void* d_out, int out_size, void* d_ws, size_t ws_size,
                              hipStream_t stream) {
  (void)in_sizes; (void)n_in; (void)out_size; (void)ws_size;
  const float* x    = (const float*)d_in[0];
  // d_in[1] = mask (all true) — unused
  const float* ln1g = (const float*)d_in[2];
  const float* ln1b = (const float*)d_in[3];
  const float* ln2g = (const float*)d_in[4];
  const float* ln2b = (const float*)d_in[5];
  const float* Wq   = (const float*)d_in[6];
  const float* bq   = (const float*)d_in[7];
  const float* Wk   = (const float*)d_in[8];
  const float* bk   = (const float*)d_in[9];
  const float* Wv   = (const float*)d_in[10];
  const float* bv   = (const float*)d_in[11];
  const float* gW   = (const float*)d_in[12];
  const float* gb   = (const float*)d_in[13];
  const float* W1   = (const float*)d_in[14];
  const float* b1   = (const float*)d_in[15];
  const float* W2   = (const float*)d_in[16];
  const float* b2   = (const float*)d_in[17];
  float* out = (float*)d_out;

  float* ws = (float*)d_ws;
  float* ln    = ws;  ws += (size_t)N_TOK * D_MODEL;
  float* q     = ws;  ws += (size_t)N_TOK * D_MODEL;
  float* k     = ws;  ws += (size_t)N_TOK * D_MODEL;
  float* v     = ws;  ws += (size_t)N_TOK * D_MODEL;
  float* o     = ws;  ws += (size_t)N_TOK * D_MODEL;
  float* xres  = ws;  ws += (size_t)N_TOK * D_MODEL;
  __bf16* hbuf = (__bf16*)ws;  ws += (size_t)N_TOK * D_FF / 2;   // bf16 hidden
  float* probs = ws;  ws += (size_t)N_TOK * N_EXP;
  float* mval  = ws;  ws += N_TOK;
  float* pi    = ws;  ws += N_TOK;
  int* eidx    = (int*)ws;    ws += N_TOK;
  int* counts  = (int*)ws;    ws += N_EXP;
  int* basep   = (int*)ws;    ws += N_EXP;
  int* cursor  = (int*)ws;    ws += N_EXP;
  int* perm    = (int*)ws;    ws += N_TOK;

  ln_kernel<<<N_TOK, 256, 0, stream>>>(x, ln1g, ln1b, ln);

  dim3 gqkv(D_MODEL / 64, N_TOK / 128);
  gemm_bias_kernel<<<gqkv, 256, 0, stream>>>(ln, Wq, bq, q, N_TOK, D_MODEL, D_MODEL);
  gemm_bias_kernel<<<gqkv, 256, 0, stream>>>(ln, Wk, bk, k, N_TOK, D_MODEL, D_MODEL);
  gemm_bias_kernel<<<gqkv, 256, 0, stream>>>(ln, Wv, bv, v, N_TOK, D_MODEL, D_MODEL);

  attn_kernel<<<dim3(SEQ / 64, BATCH * N_HEAD), 128, 0, stream>>>(q, k, v, o);
  add_kernel<<<(N_TOK * D_MODEL) / (256 * 4), 256, 0, stream>>>(x, o, xres);

  ln_kernel<<<N_TOK, 256, 0, stream>>>(xres, ln2g, ln2b, ln);

  zero_small_kernel<<<1, 32, 0, stream>>>(counts, cursor);
  gate_kernel<<<N_TOK / 8, 256, 0, stream>>>(ln, gW, gb, probs, mval, pi, eidx, counts);
  scan_kernel<<<1, 32, 0, stream>>>(counts, basep);
  perm_kernel<<<N_TOK / 256, 256, 0, stream>>>(eidx, basep, cursor, perm);

  moe_ffn1_kernel<<<dim3(D_FF / 64, N_EXP * 32), 256, 0, stream>>>(
      ln, W1, b1, perm, basep, counts, hbuf);
  moe_ffn2_kernel<<<dim3(D_MODEL / 64, N_EXP * 32), 256, 0, stream>>>(
      hbuf, W2, b2, perm, basep, counts, xres, mval, out);

  lbl_kernel<<<1, 32, 0, stream>>>(counts, pi, out);
}
